// AutoEncoder_68934225101489
// MI455X (gfx1250) — matrix operand order repaired and checked by reading, not compiled, not run on hardware
//
#include <hip/hip_runtime.h>
#include <hip/hip_bf16.h>

typedef __attribute__((ext_vector_type(16))) _Float16 v16h;
typedef __attribute__((ext_vector_type(8)))  _Float16 v8h;
typedef __attribute__((ext_vector_type(8)))  float    v8f;

#define GRU_B   8
#define GRU_T   512
#define GRU_T1  511
#define GRU_V   32000
#define GRU_D   256
#define GRU_H   256
#define ROWS    (GRU_T1 * GRU_B)   // 4088
#define ROWS_PAD 4096

// ---------------------------------------------------------------------------
// fp32 -> f16 weight conversion
// ---------------------------------------------------------------------------
__global__ void f32_to_f16_kernel(const float* __restrict__ src,
                                  _Float16* __restrict__ dst, int n) {
  int i = blockIdx.x * blockDim.x + threadIdx.x;
  if (i < n) dst[i] = (_Float16)src[i];
}

// ---------------------------------------------------------------------------
// Gather embeddings for the decode inputs: row = t*8 + b,
// token = (t==0) ? START(0) : seqs[b][t-1]; store as f16.
// ---------------------------------------------------------------------------
__global__ void gather_embed_kernel(const int* __restrict__ seqs,
                                    const float* __restrict__ emb,
                                    _Float16* __restrict__ X) {
  int row = blockIdx.x;               // 0..4087
  int t = row >> 3, b = row & 7;
  int tok = (t == 0) ? 0 : seqs[b * GRU_T + (t - 1)];
  const float* e = emb + (size_t)tok * GRU_D;
  X[(size_t)row * GRU_D + threadIdx.x] = (_Float16)e[threadIdx.x];
}

// ---------------------------------------------------------------------------
// WMMA GEMM:  C = A @ Bw^T + bias,  K = 256 fixed.
//   A  : [ROWS_PAD x 256] f16 row-major, rows >= M_actual read as zero
//   Bw : [N x 256] f16 row-major (weight matrix, we multiply by its transpose)
//   C  : indexed as (row&7)*sB + (row>>3)*sT + col  (handles the [B,T-1,V]
//        transpose of the output for free; identity for sB=768,sT=6144)
// Each wave: one 16-row M tile x eight 16-col N tiles (16x128 of C),
// A fragments held in VGPRs, B streamed from L2. Block = 8 waves stacked in M
// -> block covers 128x128 of C. Grid = (N/128, ROWS_PAD/128).
// ---------------------------------------------------------------------------
template <bool NTSTORE>
__global__ __launch_bounds__(256)
void wmma_gemm_bias(const _Float16* __restrict__ A,
                    const _Float16* __restrict__ Bw,
                    const float* __restrict__ bias,
                    float* __restrict__ C,
                    int M_actual, int N, long long sB, long long sT) {
  const int lane = threadIdx.x & 31;
  const int wave = threadIdx.x >> 5;
  const int l15  = lane & 15;
  const int half = lane >> 4;
  const int m0 = (blockIdx.y * 8 + wave) * 16;
  const int n0 = blockIdx.x * 128;

  // --- A fragments (8 x K=32 chunks), contiguous K order (k = 16*half + s), same as B ---
  v16h zero16 = {};
  v16h a[8];
  const int arow = m0 + l15;
  const bool avalid = (arow < M_actual);
  const _Float16* Ar = A + (size_t)arow * 256;
#pragma unroll
  for (int kk = 0; kk < 8; ++kk) {
    if (avalid) {
      union { v16h v; v8h h[2]; } u;
      const int kb = kk * 32 + half * 16;
      u.h[0] = *(const v8h*)(Ar + kb);        // K = kb .. kb+7
      u.h[1] = *(const v8h*)(Ar + kb + 8);    // K = kb+8 .. kb+15
      a[kk] = u.v;
    } else {
      a[kk] = zero16;
    }
  }

#pragma unroll
  for (int nt = 0; nt < 8; ++nt) {
    const int col = n0 + nt * 16 + l15;
    const _Float16* Br = Bw + (size_t)col * 256 + half * 16;
    v8f acc = {};
#pragma unroll
    for (int kk = 0; kk < 8; ++kk) {
      v16h bf = *(const v16h*)(Br + kk * 32);   // 16 contiguous K values
      acc = __builtin_amdgcn_wmma_f32_16x16x32_f16(
          false, a[kk], false, bf, (short)0, acc, false, false);
    }
    const float bv = bias[col];
#pragma unroll
    for (int r = 0; r < 8; ++r) {
      const int row = m0 + r + half * 8;
      if (row < M_actual) {
        long long idx = (long long)(row & 7) * sB + (long long)(row >> 3) * sT + col;
        float v = acc[r] + bv;
        if (NTSTORE) __builtin_nontemporal_store(v, &C[idx]);
        else         C[idx] = v;
      }
    }
  }
}

// ---------------------------------------------------------------------------
// Sequential GRU recurrence: one persistent workgroup of 12 waves (384 thr),
// 3 waves/SIMD -> per-wave VGPR budget ~341. Each wave owns FOUR 16x16
// N-tiles of gh (48 total) and keeps all its W_hh B-fragments in registers:
// 4*8*8 = 256 VGPRs + 32 accumulator + transient A ≈ 320, fits with slack
// (the 2-tile/24-wave split sat at a ~170 budget and spilled to scratch).
// b_hh staged in LDS so no bias values occupy long-lived VGPRs.
// Zero global/scratch traffic on the sequential critical path.
// ---------------------------------------------------------------------------
__global__ __launch_bounds__(384, 1)
void gru_recurrence_kernel(const float* __restrict__ GX,      // [4088 x 768], has b_ih
                           const _Float16* __restrict__ Whh,  // [768 x 256] f16
                           const float* __restrict__ bhh,     // [768]
                           _Float16* __restrict__ Hout) {     // [4088 x 256] f16
  __shared__ __attribute__((aligned(32))) _Float16 hA[16 * 256]; // 8 KB
  __shared__ float hF[8 * 256];                                  // 8 KB
  __shared__ float gh[8 * 768];                                  // 24 KB
  __shared__ float bhhS[768];                                    // 3 KB

  const int tid  = threadIdx.x;
  const int lane = tid & 31;
  const int wave = tid >> 5;          // 0..11
  const int l15  = lane & 15;
  const int half = lane >> 4;
  const int nt0  = wave * 4;          // this wave's four N-tiles

  // --- preload this wave's W_hh fragments into registers (live all 511 steps)
  v16h bf[4][8];
#pragma unroll
  for (int q = 0; q < 4; ++q) {
    const _Float16* Br = Whh + (size_t)((nt0 + q) * 16 + l15) * 256 + half * 16;
#pragma unroll
    for (int kk = 0; kk < 8; ++kk) bf[q][kk] = *(const v16h*)(Br + kk * 32);
  }

  for (int i = tid; i < 768; i += 384)      bhhS[i] = bhh[i];
  for (int i = tid; i < 16 * 256; i += 384) hA[i] = (_Float16)0.f;
  for (int i = tid; i < 8 * 256; i += 384)  hF[i] = 0.f;
  __syncthreads();

  for (int t = 0; t < GRU_T1; ++t) {
    // ---- gh = h @ Whh^T : four independent accumulator chains per wave ----
    v8f acc[4] = {};
    {
      const _Float16* Ar = hA + l15 * 256;
#pragma unroll
      for (int kk = 0; kk < 8; ++kk) {
        union { v16h v; v8h h[2]; } u;   // transient A fragment
        const int kb = kk * 32 + half * 16;
        u.h[0] = *(const v8h*)(Ar + kb);
        u.h[1] = *(const v8h*)(Ar + kb + 8);
#pragma unroll
        for (int q = 0; q < 4; ++q) {
          acc[q] = __builtin_amdgcn_wmma_f32_16x16x32_f16(
              false, u.v, false, bf[q][kk], (short)0, acc[q], false, false);
        }
      }
    }
    if (half == 0) {   // rows 0..7 are the real batch rows
#pragma unroll
      for (int q = 0; q < 4; ++q)
#pragma unroll
        for (int r = 0; r < 8; ++r)
          gh[r * 768 + (nt0 + q) * 16 + l15] = acc[q][r];
    }
    __syncthreads();

    // ---- GRU elementwise update: 2048 elements over 384 threads ----
    const float* gx = GX + (size_t)t * (8 * 768);
    for (int e = tid; e < 8 * 256; e += 384) {
      const int b = e >> 8, j = e & 255;
      const float* g = gx + b * 768;
      const float xr = g[j], xz = g[256 + j], xn = g[512 + j];
      const float hr = gh[b * 768 + j]       + bhhS[j];
      const float hz = gh[b * 768 + 256 + j] + bhhS[256 + j];
      const float hn = gh[b * 768 + 512 + j] + bhhS[512 + j];
      const float r  = 1.f / (1.f + __expf(-(xr + hr)));
      const float z  = 1.f / (1.f + __expf(-(xz + hz)));
      const float nn = tanhf(xn + r * hn);
      const float hnew = (1.f - z) * nn + z * hF[e];
      hF[e] = hnew;
      const _Float16 h16 = (_Float16)hnew;
      hA[b * 256 + j] = h16;
      Hout[(size_t)(t * 8 + b) * 256 + j] = h16;
    }
    __syncthreads();
  }
}

// ---------------------------------------------------------------------------
// Launch
// ---------------------------------------------------------------------------
extern "C" void kernel_launch(void* const* d_in, const int* in_sizes, int n_in,
                              void* d_out, int out_size, void* d_ws, size_t ws_size,
                              hipStream_t stream) {
  // inputs: labels, seqs, emb, W_ih, W_hh, b_ih, b_hh, W_out, b_out
  const int*   seqs  = (const int*)  d_in[1];
  const float* emb   = (const float*)d_in[2];
  const float* W_ih  = (const float*)d_in[3];
  const float* W_hh  = (const float*)d_in[4];
  const float* b_ih  = (const float*)d_in[5];
  const float* b_hh  = (const float*)d_in[6];
  const float* W_out = (const float*)d_in[7];
  const float* b_out = (const float*)d_in[8];
  float* out = (float*)d_out;

  // workspace layout (all offsets 32B-aligned), ~34 MB total
  char* w = (char*)d_ws;
  _Float16* Xf16   = (_Float16*)(w + 0);          // 4096*256*2  = 2,097,152
  _Float16* Wih16  = (_Float16*)(w + 2097152);    //  768*256*2  =   393,216
  _Float16* Whh16  = (_Float16*)(w + 2490368);    //  768*256*2  =   393,216
  _Float16* Wout16 = (_Float16*)(w + 2883584);    // 32000*256*2 = 16,384,000
  float*    GX     = (float*)   (w + 19267584);   // 4096*768*4  = 12,582,912
  _Float16* Hf16   = (_Float16*)(w + 31850496);   // 4096*256*2  = 2,097,152

  // 1) weights -> f16
  f32_to_f16_kernel<<<(196608  + 255) / 256, 256, 0, stream>>>(W_ih,  Wih16,  196608);
  f32_to_f16_kernel<<<(196608  + 255) / 256, 256, 0, stream>>>(W_hh,  Whh16,  196608);
  f32_to_f16_kernel<<<(8192000 + 255) / 256, 256, 0, stream>>>(W_out, Wout16, 8192000);

  // 2) gather decode-input embeddings (f16)
  gather_embed_kernel<<<ROWS, 256, 0, stream>>>(seqs, emb, Xf16);

  // 3) GX = X @ W_ih^T + b_ih   (identity row mapping: sB=768, sT=8*768)
  wmma_gemm_bias<false><<<dim3(768 / 128, ROWS_PAD / 128), 256, 0, stream>>>(
      Xf16, Wih16, b_ih, GX, ROWS, 768, 768LL, 8LL * 768LL);

  // 4) sequential GRU recurrence -> H (f16), W_hh register-resident
  gru_recurrence_kernel<<<1, 384, 0, stream>>>(GX, Whh16, b_hh, Hf16);

  // 5) logits = H @ W_out^T + b_out, written as [B, T-1, V] via row remap,
  //    nontemporal stores (523 MB streamed once, never re-read)
  wmma_gemm_bias<true><<<dim3(GRU_V / 128, ROWS_PAD / 128), 256, 0, stream>>>(
      Hf16, Wout16, b_out, out, ROWS, GRU_V,
      (long long)GRU_T1 * GRU_V, (long long)GRU_V);
}
